// C4WiredTransformer_62380105007242
// MI455X (gfx1250) — compile-verified, matrix operand order checked by reading, not run on hardware
//
#include <hip/hip_runtime.h>
#include <hip/hip_bf16.h>

#define EMU_MEM_SIZE 33554432LL

typedef float    v2f  __attribute__((ext_vector_type(2)));
typedef float    v8f  __attribute__((ext_vector_type(8)));
typedef _Float16 v16h __attribute__((ext_vector_type(16)));

// ---------------------------------------------------------------------------
// Big passthrough: memory (int32 on device per harness ABI) -> float output.
// 16B loads / 16B stores per lane; this is the bandwidth-bound part
// (~268 MB total -> ~11.5us at 23.3 TB/s).
// ---------------------------------------------------------------------------
__global__ void emu_copy_mem_kernel(const int* __restrict__ mem,
                                    float* __restrict__ out, int n) {
    int i = (int)(blockIdx.x * blockDim.x + threadIdx.x) * 4;
    if (i + 3 < n) {
        int4 v = *reinterpret_cast<const int4*>(mem + i);
        float4 f;
        f.x = (float)v.x; f.y = (float)v.y; f.z = (float)v.z; f.w = (float)v.w;
        *reinterpret_cast<float4*>(out + i) = f;
    } else {
        for (int j = i; j < n; ++j) out[j] = (float)mem[j];
    }
}

// ---------------------------------------------------------------------------
// Helpers for the single-wave emulator step
// ---------------------------------------------------------------------------
__device__ __forceinline__ long long emu_clamp(long long a) {
    if (a < 0) return 0;
    if (a > EMU_MEM_SIZE - 1) return EMU_MEM_SIZE - 1;
    return a;
}

__device__ __forceinline__ long long emu_read_int(const int* __restrict__ mem,
                                                  long long addr) {
    long long v = 0;
#pragma unroll
    for (int i = 0; i < 8; ++i)
        v += ((long long)mem[emu_clamp(addr + i)]) << (8 * i);
    return v;
}

__device__ __forceinline__ long long emu_fdiv(long long a, long long b) {
    long long q = a / b;
    if ((a % b != 0) && ((a < 0) != (b < 0))) --q;   // Python floor division
    return q;
}
__device__ __forceinline__ long long emu_fmod(long long a, long long b) {
    long long r = a % b;
    if (r != 0 && ((r < 0) != (b < 0))) r += b;      // Python modulo
    return r;
}

// Guarded weight fetch (zero-pads both K and N past the true matrix bounds)
__device__ __forceinline__ float emu_welt(const float* __restrict__ W,
                                          int nrows, int ncols, int k, int col) {
    return (k < nrows && col < ncols) ? W[k * ncols + col] : 0.0f;
}

// Compute row 0 of D = vec(1xK, zero-padded in LDS) * W[K x ncols] tile at
// columns [col0, col0+16) using WMMA. Result row lands in D VGPR0 lanes 0-15.
__device__ v8f emu_wmma_rowvec(const float* vec, int K,
                               const float* __restrict__ W,
                               int nrows, int ncols, int col0, int lane) {
    v8f acc = {0.f, 0.f, 0.f, 0.f, 0.f, 0.f, 0.f, 0.f};
    const bool hi = lane >= 16;
    const int  n  = lane & 15;
#if __has_builtin(__builtin_amdgcn_wmma_f32_16x16x4_f32)
    // A 16x4 f32: lanes 0-15 -> M=lane, VGPR0=K0/VGPR1=K1; lanes 16-31 -> K2/K3.
    // Only row M=0 (lanes 0 and 16) carries the vector; all other rows zero.
    const int chunks = (K + 3) >> 2;
    for (int c = 0; c < chunks; ++c) {
        v2f a;
        a.x = (lane == 0) ? vec[4 * c + 0] : ((lane == 16) ? vec[4 * c + 2] : 0.0f);
        a.y = (lane == 0) ? vec[4 * c + 1] : ((lane == 16) ? vec[4 * c + 3] : 0.0f);
        const int kb = 4 * c + (hi ? 2 : 0);
        v2f b;
        b.x = emu_welt(W, nrows, ncols, kb,     col0 + n);
        b.y = emu_welt(W, nrows, ncols, kb + 1, col0 + n);
        acc = __builtin_amdgcn_wmma_f32_16x16x4_f32(
            false, a, false, b, (short)0, acc, false, false);
    }
#else
    // Fallback: codegen-confirmed f32<-f16 16x16x32 WMMA, same row-0 trick.
    const int chunks = (K + 31) >> 5;
    for (int c = 0; c < chunks; ++c) {
        v16h a, b;
#pragma unroll
        for (int j = 0; j < 16; ++j) {
            const int klocal = (hi ? 8 : 0) + (j & 7) + ((j >= 8) ? 16 : 0);
            const int k = 32 * c + klocal;
            float av = 0.0f;
            if (lane == 0 || lane == 16) av = vec[k];
            a[j] = (_Float16)av;
            b[j] = (_Float16)emu_welt(W, nrows, ncols, k, col0 + n);
        }
        acc = __builtin_amdgcn_wmma_f32_16x16x32_f16(
            false, a, false, b, (short)0, acc, false, false);
    }
#endif
    return acc;
}

// ---------------------------------------------------------------------------
// Single-wave emulator step: MLP via WMMA + exact int64 semantics + scatter.
// Runs after the copy kernel on the same stream, so scatter writes land on
// top of the copied memory image.
// ---------------------------------------------------------------------------
__global__ void __launch_bounds__(32)
emu_step_kernel(const int* __restrict__ pc_p, const int* __restrict__ sp_p,
                const int* __restrict__ bp_p, const int* __restrict__ ax_p,
                const int* __restrict__ mem,
                const float* __restrict__ W1, const float* __restrict__ b1,
                const float* __restrict__ W2, const float* __restrict__ b2,
                float* __restrict__ out) {
    __shared__ float xbuf[32];     // x padded with zeros to K-chunk multiple
    __shared__ float hbuf[64];     // hidden layer
    __shared__ float outbuf[160];  // 156 outputs padded to 160
    const int lane = threadIdx.x;

    xbuf[lane] = 0.0f;
    hbuf[lane] = 0.0f;
    hbuf[lane + 32] = 0.0f;
    for (int i = lane; i < 160; i += 32) outbuf[i] = 0.0f;

    long long pc = 0, sp = 0, bp = 0, ax = 0;
    long long instruction = 0, imm = 0, stack_top = 0;
    long long mem_ax = 0, mem_bp = 0, mem_bp8 = 0;
    if (lane == 0) {
        pc = (long long)pc_p[0]; sp = (long long)sp_p[0];
        bp = (long long)bp_p[0]; ax = (long long)ax_p[0];
        instruction = emu_read_int(mem, pc);
        imm         = instruction >> 8;
        stack_top   = emu_read_int(mem, sp);
        mem_ax      = emu_read_int(mem, ax);
        mem_bp      = emu_read_int(mem, bp);
        mem_bp8     = emu_read_int(mem, bp + 8);
        xbuf[0] = (float)pc;        xbuf[1] = (float)sp;
        xbuf[2] = (float)bp;        xbuf[3] = (float)ax;
        xbuf[4] = (float)imm;       xbuf[5] = (float)stack_top;
        xbuf[6] = (float)mem_ax;    xbuf[7] = (float)mem_bp;
        xbuf[8] = (float)mem_bp8;
    }
    __syncthreads();

    // Layer 1: h = x @ W1 + b1   (1x9 * 9x64, 4 N-tiles)
    for (int t = 0; t < 4; ++t) {
        v8f acc = emu_wmma_rowvec(xbuf, 9, W1, 9, 64, 16 * t, lane);
        if (lane < 16) hbuf[16 * t + lane] = acc[0] + b1[16 * t + lane];
    }
    __syncthreads();

    // Layer 2: out = h @ W2 + b2  (1x64 * 64x156, 10 N-tiles)
    for (int t = 0; t < 10; ++t) {
        v8f acc = emu_wmma_rowvec(hbuf, 64, W2, 64, 156, 16 * t, lane);
        const int col = 16 * t + lane;
        if (lane < 16 && col < 156) outbuf[col] = acc[0] + b2[col];
    }
    __syncthreads();

    if (lane == 0) {
        const long long opcode = instruction & 255;
        // softmax(-(opcode-k)^2 / 0.01) soft-select over 39 rows of 4
        float attn[39];
        float mx = -3.4e38f;
        const float ocf = (float)opcode;
        for (int k = 0; k < 39; ++k) {
            const float d = ocf - (float)k;
            const float l = -(d * d) * 100.0f;
            attn[k] = l;
            if (l > mx) mx = l;
        }
        float s = 0.0f;
        for (int k = 0; k < 39; ++k) { attn[k] = __expf(attn[k] - mx); s += attn[k]; }
        const float inv = 1.0f / s;
        float sel[4] = {0.f, 0.f, 0.f, 0.f};
        for (int k = 0; k < 39; ++k) {
            const float w = attn[k] * inv;
            sel[0] += w * outbuf[4 * k + 0];
            sel[1] += w * outbuf[4 * k + 1];
            sel[2] += w * outbuf[4 * k + 2];
            sel[3] += w * outbuf[4 * k + 3];
        }
        long long new_pc = (long long)sel[0];
        long long new_sp = (long long)sel[1];
        long long new_bp = (long long)sel[2];
        long long new_ax = (long long)sel[3];

        const long long div_safe = (ax == 0) ? 1LL : ax;
        switch ((int)opcode) {
            case 16: new_ax = stack_top * ax; break;
            case 17: new_ax = emu_fdiv(stack_top, div_safe); break;
            case 18: new_ax = emu_fmod(stack_top, div_safe); break;
            case 19: new_ax = stack_top | ax; break;
            case 20: new_ax = stack_top ^ ax; break;
            case 21: new_ax = stack_top & ax; break;
            case 22: new_ax = (long long)((unsigned long long)stack_top << (int)(ax & 63)); break;
            case 23: new_ax = stack_top >> (int)(ax & 63); break;   // arithmetic
            case 24: new_ax = (long long)(stack_top == ax); break;
            case 25: new_ax = (long long)(stack_top != ax); break;
            case 26: new_ax = (long long)(stack_top <  ax); break;
            case 27: new_ax = (long long)(stack_top >  ax); break;
            case 28: new_ax = (long long)(stack_top <= ax); break;
            case 29: new_ax = (long long)(stack_top >= ax); break;
            case 10: new_ax = mem_ax & 255; break;
            default: break;
        }
        if (opcode == 4) new_pc = (ax == 0) ? imm : pc + 8;   // BZ
        if (opcode == 5) new_pc = (ax != 0) ? imm : pc + 8;   // BNZ
        if (opcode == 7) new_sp = sp + imm;                    // ADJ

        out[0] = (float)new_pc;
        out[1] = (float)new_sp;
        out[2] = (float)new_bp;
        out[3] = (float)new_ax;

        // masked scatter writes on top of the copied memory image
        float* om = out + 4;
        if (opcode == 13 || opcode == 3 || opcode == 6) {     // PSH / JSR / ENT
            const long long wv = (opcode == 13) ? ax
                               : (opcode == 3)  ? (pc + 8) : bp;
            for (int i = 0; i < 8; ++i)
                om[emu_clamp(sp - 8 + i)] = (float)((wv >> (8 * i)) & 255);
        }
        if (opcode == 11) {                                    // SI
            for (int i = 0; i < 8; ++i)
                om[emu_clamp(stack_top + i)] = (float)((ax >> (8 * i)) & 255);
        }
        if (opcode == 12) {                                    // SC
            om[emu_clamp(stack_top)] = (float)(ax & 255);
        }
    }
}

// ---------------------------------------------------------------------------
extern "C" void kernel_launch(void* const* d_in, const int* in_sizes, int n_in,
                              void* d_out, int out_size, void* d_ws, size_t ws_size,
                              hipStream_t stream) {
    const int*   pc  = (const int*)d_in[0];
    const int*   sp  = (const int*)d_in[1];
    const int*   bp  = (const int*)d_in[2];
    const int*   ax  = (const int*)d_in[3];
    const int*   mem = (const int*)d_in[4];
    const float* W1  = (const float*)d_in[5];
    const float* b1  = (const float*)d_in[6];
    const float* W2  = (const float*)d_in[7];
    const float* b2  = (const float*)d_in[8];
    float* out = (float*)d_out;

    const int n = in_sizes[4];                 // 33554432 memory elements
    const int threads = 256;
    const int per = 4;                          // elements per lane (16B ld/st)
    const int blocks = (n + threads * per - 1) / (threads * per);

    emu_copy_mem_kernel<<<blocks, threads, 0, stream>>>(mem, out + 4, n);
    emu_step_kernel<<<1, 32, 0, stream>>>(pc, sp, bp, ax, mem,
                                          W1, b1, W2, b2, out);
}
